// ieHGCNConv_27118423507478
// MI455X (gfx1250) — matrix operand order repaired and checked by reading, not compiled, not run on hardware
//
#include <hip/hip_runtime.h>
#include <hip/hip_bf16.h>

typedef __attribute__((ext_vector_type(16))) _Float16 v16h;
typedef __attribute__((ext_vector_type(8)))  _Float16 v8h;
typedef __attribute__((ext_vector_type(8)))  float    v8f;

#define NA 40000
#define NP 80000
#define NE 800000
#define DD 256

// ---------------------------------------------------------------- utilities
__global__ void k_zero(float* __restrict__ p, long n) {
    long i = (long)blockIdx.x * blockDim.x + threadIdx.x;
    if (i < n) p[i] = 0.0f;
}

__global__ void k_f32_to_f16(const float* __restrict__ in, _Float16* __restrict__ out, long n) {
    long i = (long)blockIdx.x * blockDim.x + threadIdx.x;
    if (i < n) out[i] = (_Float16)in[i];
}

// WT[n][k] = W[k][n], f32 -> f16 (weight is 256x256)
__global__ void k_wtrans(const float* __restrict__ W, _Float16* __restrict__ WT) {
    int n = blockIdx.x, k = threadIdx.x;
    WT[n * DD + k] = (_Float16)W[k * DD + n];
}

// Fold attention projections: uq[j] = sum_t Wq[j][t]*w_ar[t]; uk[j] = sum_t Wk[j][t]*w_al[t]
// c2[0] = bq.w_ar + b_ar ; c2[1] = bk.w_al + b_al
__global__ void k_attn_prep(const float* __restrict__ Wq, const float* __restrict__ bq,
                            const float* __restrict__ Wk, const float* __restrict__ bk,
                            const float* __restrict__ w_al, const float* __restrict__ b_al,
                            const float* __restrict__ w_ar, const float* __restrict__ b_ar,
                            float* __restrict__ uq, float* __restrict__ uk, float* __restrict__ c2) {
    int j = threadIdx.x;  // 0..255
    float sq = 0.0f, sk = 0.0f;
    for (int t = 0; t < 64; ++t) {
        sq += Wq[j * 64 + t] * w_ar[t];
        sk += Wk[j * 64 + t] * w_al[t];
    }
    uq[j] = sq;
    uk[j] = sk;
    if (j == 0) { float s = 0.0f; for (int t = 0; t < 64; ++t) s += bq[t] * w_ar[t]; c2[0] = s + b_ar[0]; }
    if (j == 1) { float s = 0.0f; for (int t = 0; t < 64; ++t) s += bk[t] * w_al[t]; c2[1] = s + b_al[0]; }
}

// ---------------------------------------------------------------- graph aggregation
__global__ void k_deg(const int* __restrict__ dst, float* __restrict__ deg, int e) {
    int i = blockIdx.x * blockDim.x + threadIdx.x;
    if (i < e) atomicAdd(&deg[dst[i]], 1.0f);
}

// one block (256 threads) per edge; lane c moves column c
__global__ void k_scatter(const float* __restrict__ h, const int* __restrict__ src,
                          const int* __restrict__ dst, float* __restrict__ agg) {
    int e = blockIdx.x;
    int c = threadIdx.x;
    int s = src[e], d = dst[e];
    atomicAdd(&agg[(long)d * DD + c], h[(long)s * DD + c]);
}

// normalized aggregate -> f16
__global__ void k_norm_f16(const float* __restrict__ agg, const float* __restrict__ deg,
                           _Float16* __restrict__ out, long n) {
    long i = (long)blockIdx.x * blockDim.x + threadIdx.x;
    if (i < n) {
        float d = deg[i >> 8];
        d = d < 1.0f ? 1.0f : d;
        out[i] = (_Float16)(agg[i] / d);
    }
}

// ---------------------------------------------------------------- WMMA GEMM
// out[m][n] = sum_k A[m][k] * W[k][n] + bias[n], A f16 [rows][256], WT f16 [256(n)][256(k)]
// grid.x = rows/16, block = 128 (4 waves); wave w computes n-cols [w*64, w*64+64)
__global__ __launch_bounds__(128)
void k_gemm_wmma(const _Float16* __restrict__ A, const _Float16* __restrict__ WT,
                 const float* __restrict__ bias, float* __restrict__ out) {
    const int m0   = blockIdx.x * 16;
    const int lane = threadIdx.x & 31;
    const int wave = threadIdx.x >> 5;
    const int n0   = wave * 64;
    const int l16  = lane & 15;
    const int hi   = lane >> 4;   // 0: lanes 0-15, 1: lanes 16-31

    v8f acc[4] = {};

    const _Float16* arow = A + (long)(m0 + l16) * DD;

    for (int kb = 0; kb < DD; kb += 32) {
        if (kb + 32 < DD) __builtin_prefetch(arow + kb + 32, 0, 1);
        // A fragment (contiguous K, matching B): lanes 0-15 K {kb..kb+15},
        // lanes 16-31 K {kb+16..kb+31}
        v8h alo = *(const v8h*)(arow + kb + hi * 16);
        v8h ahi = *(const v8h*)(arow + kb + hi * 16 + 8);
        v16h a;
#pragma unroll
        for (int i = 0; i < 8; ++i) { a[i] = alo[i]; a[i + 8] = ahi[i]; }

#pragma unroll
        for (int j = 0; j < 4; ++j) {
            // B fragment: lane holds column n0+j*16+l16, K kb+hi*16 .. +16 contiguous in WT
            const _Float16* wrow = WT + (long)(n0 + j * 16 + l16) * DD + kb + hi * 16;
            v16h b = *(const v16h*)wrow;
            acc[j] = __builtin_amdgcn_wmma_f32_16x16x32_f16(
                false, a, false, b, (short)0, acc[j], false, false);
        }
    }

#pragma unroll
    for (int j = 0; j < 4; ++j) {
        int col = n0 + j * 16 + l16;
        float bv = bias[col];
#pragma unroll
        for (int r = 0; r < 8; ++r) {
            // C/D layout: VGPR r -> M = m0 + hi*8 + r, N = col
            out[(long)(m0 + hi * 8 + r) * DD + col] = acc[j][r] + bv;
        }
    }
}

// ---------------------------------------------------------------- attention combine
__device__ __forceinline__ float eluf(float x) { return x > 0.0f ? x : __expf(x) - 1.0f; }

__device__ __forceinline__ float wred(float v) {
#pragma unroll
    for (int m = 16; m >= 1; m >>= 1) v += __shfl_xor(v, m, 32);
    return v;
}

// one wave32 per node; lane handles 8 columns
__global__ __launch_bounds__(256)
void k_combine(const float* __restrict__ z, const float* __restrict__ cv,
               const float* __restrict__ uq, const float* __restrict__ uk,
               const float* __restrict__ c2, float* __restrict__ out, int n) {
    int node = blockIdx.x * 8 + (threadIdx.x >> 5);
    int lane = threadIdx.x & 31;
    if (node >= n) return;

    const float* zp = z  + (long)node * DD + lane * 8;
    const float* cp = cv + (long)node * DD + lane * 8;
    float zr[8], cr[8], uqv[8], ukv[8];
#pragma unroll
    for (int i = 0; i < 8; ++i) {
        zr[i]  = zp[i];
        cr[i]  = cp[i];
        uqv[i] = uq[lane * 8 + i];
        ukv[i] = uk[lane * 8 + i];
    }
    float ph = 0.0f, ps = 0.0f, pc = 0.0f;
#pragma unroll
    for (int i = 0; i < 8; ++i) {
        ph += zr[i] * uqv[i];
        ps += zr[i] * ukv[i];
        pc += cr[i] * ukv[i];
    }
    ph = wred(ph); ps = wred(ps); pc = wred(pc);

    float hr = ph + c2[0];
    float ls = eluf(ps + c2[1] + hr);
    float lc = eluf(pc + c2[1] + hr);
    float mx = fmaxf(ls, lc);
    float e0 = __expf(ls - mx), e1 = __expf(lc - mx);
    float inv = 1.0f / (e0 + e1);
    float a0 = e0 * inv, a1 = e1 * inv;

    float* op = out + (long)node * DD + lane * 8;
#pragma unroll
    for (int i = 0; i < 8; ++i) op[i] = eluf(a0 * zr[i] + a1 * cr[i]);
}

// ---------------------------------------------------------------- launcher
extern "C" void kernel_launch(void* const* d_in, const int* in_sizes, int n_in,
                              void* d_out, int out_size, void* d_ws, size_t ws_size,
                              hipStream_t stream) {
    (void)in_sizes; (void)n_in; (void)out_size; (void)ws_size;

    const float* h_author = (const float*)d_in[0];
    const float* h_paper  = (const float*)d_in[1];
    const int*   src_ap   = (const int*)d_in[2];
    const int*   dst_ap   = (const int*)d_in[3];
    const int*   src_pa   = (const int*)d_in[4];
    const int*   dst_pa   = (const int*)d_in[5];
    const float* W_self_a = (const float*)d_in[6];  const float* b_self_a = (const float*)d_in[7];
    const float* W_self_p = (const float*)d_in[8];  const float* b_self_p = (const float*)d_in[9];
    const float* Wq_a = (const float*)d_in[10]; const float* bq_a = (const float*)d_in[11];
    const float* Wq_p = (const float*)d_in[12]; const float* bq_p = (const float*)d_in[13];
    const float* Wk_a = (const float*)d_in[14]; const float* bk_a = (const float*)d_in[15];
    const float* Wk_p = (const float*)d_in[16]; const float* bk_p = (const float*)d_in[17];
    const float* w_al_a = (const float*)d_in[18]; const float* b_al_a = (const float*)d_in[19];
    const float* w_al_p = (const float*)d_in[20]; const float* b_al_p = (const float*)d_in[21];
    const float* w_ar_a = (const float*)d_in[22]; const float* b_ar_a = (const float*)d_in[23];
    const float* w_ar_p = (const float*)d_in[24]; const float* b_ar_p = (const float*)d_in[25];
    const float* W_ap = (const float*)d_in[26]; const float* b_ap = (const float*)d_in[27];
    const float* W_pa = (const float*)d_in[28]; const float* b_pa = (const float*)d_in[29];

    char* ws = (char*)d_ws;
    // workspace layout (agg_a, agg_p, deg_a, deg_p contiguous so one zero-pass covers them)
    size_t off = 0;
    float*    aggA = (float*)(ws + off);    off += (size_t)NA * DD * 4;   // later reused as conv_a
    float*    aggP = (float*)(ws + off);    off += (size_t)NP * DD * 4;   // later reused as conv_p
    float*    degA = (float*)(ws + off);    off += (size_t)NA * 4;
    float*    degP = (float*)(ws + off);    off += (size_t)NP * 4;
    _Float16* haH  = (_Float16*)(ws + off); off += (size_t)NA * DD * 2;
    _Float16* hpH  = (_Float16*)(ws + off); off += (size_t)NP * DD * 2;
    _Float16* aaH  = (_Float16*)(ws + off); off += (size_t)NA * DD * 2;
    _Float16* apH  = (_Float16*)(ws + off); off += (size_t)NP * DD * 2;
    float*    zA   = (float*)(ws + off);    off += (size_t)NA * DD * 4;
    float*    zP   = (float*)(ws + off);    off += (size_t)NP * DD * 4;
    _Float16* WTsa = (_Float16*)(ws + off); off += (size_t)DD * DD * 2;
    _Float16* WTsp = (_Float16*)(ws + off); off += (size_t)DD * DD * 2;
    _Float16* WTpa = (_Float16*)(ws + off); off += (size_t)DD * DD * 2;
    _Float16* WTap = (_Float16*)(ws + off); off += (size_t)DD * DD * 2;
    float*    uqA  = (float*)(ws + off);    off += DD * 4;
    float*    ukA  = (float*)(ws + off);    off += DD * 4;
    float*    uqP  = (float*)(ws + off);    off += DD * 4;
    float*    ukP  = (float*)(ws + off);    off += DD * 4;
    float*    cA   = (float*)(ws + off);    off += 8;
    float*    cP   = (float*)(ws + off);    off += 8;

    float* out_a = (float*)d_out;
    float* out_p = (float*)d_out + (long)NA * DD;

    // 1) zero aggregation buffers + degrees (contiguous)
    long zcnt = (long)(NA + NP) * DD + NA + NP;
    k_zero<<<(unsigned)((zcnt + 255) / 256), 256, 0, stream>>>(aggA, zcnt);

    // 2) convert features to f16
    long na = (long)NA * DD, np = (long)NP * DD;
    k_f32_to_f16<<<(unsigned)((na + 255) / 256), 256, 0, stream>>>(h_author, haH, na);
    k_f32_to_f16<<<(unsigned)((np + 255) / 256), 256, 0, stream>>>(h_paper,  hpH, np);

    // 3) weight prep
    k_wtrans<<<DD, DD, 0, stream>>>(W_self_a, WTsa);
    k_wtrans<<<DD, DD, 0, stream>>>(W_self_p, WTsp);
    k_wtrans<<<DD, DD, 0, stream>>>(W_pa,     WTpa);
    k_wtrans<<<DD, DD, 0, stream>>>(W_ap,     WTap);
    k_attn_prep<<<1, DD, 0, stream>>>(Wq_a, bq_a, Wk_a, bk_a, w_al_a, b_al_a, w_ar_a, b_ar_a, uqA, ukA, cA);
    k_attn_prep<<<1, DD, 0, stream>>>(Wq_p, bq_p, Wk_p, bk_p, w_al_p, b_al_p, w_ar_p, b_ar_p, uqP, ukP, cP);

    // 4) graph aggregation (paper->author feeds conv_a; author->paper feeds conv_p)
    k_deg<<<(NE + 255) / 256, 256, 0, stream>>>(dst_pa, degA, NE);
    k_deg<<<(NE + 255) / 256, 256, 0, stream>>>(dst_ap, degP, NE);
    k_scatter<<<NE, DD, 0, stream>>>(h_paper,  src_pa, dst_pa, aggA);
    k_scatter<<<NE, DD, 0, stream>>>(h_author, src_ap, dst_ap, aggP);

    // 5) normalize + convert aggregates to f16
    k_norm_f16<<<(unsigned)((na + 255) / 256), 256, 0, stream>>>(aggA, degA, aaH, na);
    k_norm_f16<<<(unsigned)((np + 255) / 256), 256, 0, stream>>>(aggP, degP, apH, np);

    // 6) WMMA GEMMs (conv outputs reuse the f32 aggregation buffers)
    k_gemm_wmma<<<NA / 16, 128, 0, stream>>>(haH, WTsa, b_self_a, zA);
    k_gemm_wmma<<<NP / 16, 128, 0, stream>>>(hpH, WTsp, b_self_p, zP);
    k_gemm_wmma<<<NA / 16, 128, 0, stream>>>(aaH, WTpa, b_pa, aggA);
    k_gemm_wmma<<<NP / 16, 128, 0, stream>>>(apH, WTap, b_ap, aggP);

    // 7) attention combine -> outputs
    k_combine<<<NA / 8, 256, 0, stream>>>(zA, aggA, uqA, ukA, cA, out_a, NA);
    k_combine<<<NP / 8, 256, 0, stream>>>(zP, aggP, uqP, ukP, cP, out_p, NP);
}
